// DotProductAttention_10256381903069
// MI455X (gfx1250) — matrix operand order repaired and checked by reading, not compiled, not run on hardware
//
#include <hip/hip_runtime.h>

typedef _Float16 v16h __attribute__((ext_vector_type(16)));
typedef _Float16 v8h  __attribute__((ext_vector_type(8)));
typedef float    v8f  __attribute__((ext_vector_type(8)));

#define SEQ 2048
#define HD  64
#define NQT (SEQ / 16)
#define KB  64            // keys per inner block
#define NKB (SEQ / KB)
#define WPB 8             // waves per block (8 q-tiles share K/V staging)

// Native v_exp_f32 (exp2)
__device__ __forceinline__ float fast_exp2(float x) {
  return __builtin_amdgcn_exp2f(x);
}

// ds_swizzle xor-16 within wave32 (SWAPX16: offset = xor<<10 | and=0x1f)
__device__ __forceinline__ float swz16(float v) {
  return __int_as_float(__builtin_amdgcn_ds_swizzle(__float_as_int(v), 0x401F));
}

__device__ __forceinline__ v8h pack8(float4 a, float4 b, float s) {
  v8h r;
  r[0] = (_Float16)(a.x * s); r[1] = (_Float16)(a.y * s);
  r[2] = (_Float16)(a.z * s); r[3] = (_Float16)(a.w * s);
  r[4] = (_Float16)(b.x * s); r[5] = (_Float16)(b.y * s);
  r[6] = (_Float16)(b.z * s); r[7] = (_Float16)(b.w * s);
  return r;
}

// Flash attention, transposed form: S^T = K * Q^T, O^T = V^T * P^T.
// 256 threads = 8 waves; each wave owns one 16-row Q tile; the block shares
// f16-staged K (row-major) and V^T tiles in LDS, double buffered.
__global__ __launch_bounds__(256)
void fa_fwd_kernel(const float* __restrict__ Qg, const float* __restrict__ Kg,
                   const float* __restrict__ Vg, float* __restrict__ Og) {
  __shared__ __align__(16) _Float16 ldsK[2][KB * HD];   // [buf][k][d]   8KB each
  __shared__ __align__(16) _Float16 ldsVT[2][HD * KB];  // [buf][d][k]   8KB each

  const int tid    = threadIdx.x;
  const int lane   = tid & 31;
  const int w      = tid >> 5;
  const int lane16 = lane & 15;
  const int hsel   = lane >> 4;

  const int bh = blockIdx.x / (NQT / WPB);
  const int qt = (blockIdx.x % (NQT / WPB)) * WPB + w;

  const size_t base = (size_t)bh * SEQ * HD;
  const float* q = Qg + base;
  const float* k = Kg + base;
  const float* v = Vg + base;
  float*       o = Og + base;

  // ---- Q^T B-fragments (two 32d x 16q tiles), scaled by log2(e)/sqrt(64) ----
  // B layout: lane holds column n = q = lane16; half e -> K(d) = 32c + 8*hsel + e(+16)
  const float QSCALE = 0.125f * 1.4426950408889634f;
  v16h bq[2];
  {
    const float* qp = q + (size_t)(qt * 16 + lane16) * HD;
#pragma unroll
    for (int c = 0; c < 2; ++c) {
      const float4* p = (const float4*)(qp + 32 * c + 8 * hsel);
      const float4 f0 = p[0], f1 = p[1], f2 = p[4], f3 = p[5];
      const v8h lo = pack8(f0, f1, QSCALE);
      const v8h hi = pack8(f2, f3, QSCALE);
#pragma unroll
      for (int e = 0; e < 8; ++e) { bq[c][e] = lo[e]; bq[c][8 + e] = hi[e]; }
    }
  }

  // Per-lane softmax state: this lane owns query column q = lane16 (exp2 domain).
  float m_run = -1.0e30f, l_run = 0.0f;
  v8f acc[4];  // O^T accumulators: lane holds q=lane16, d = 16f + r + 8*hsel
#pragma unroll
  for (int f = 0; f < 4; ++f) acc[f] = {};

  // ---- cooperative staging: thread handles K/V row srow, cols scol..scol+15 ----
  const int srow = tid >> 2;           // 0..63
  const int scol = (tid & 3) * 16;     // 0,16,32,48
  auto stage = [&](int buf, int k0) {
    const float4* kp = (const float4*)(k + (size_t)(k0 + srow) * HD + scol);
    const float4* vp = (const float4*)(v + (size_t)(k0 + srow) * HD + scol);
    const float4 k0v = kp[0], k1v = kp[1], k2v = kp[2], k3v = kp[3];
    const float4 v0v = vp[0], v1v = vp[1], v2v = vp[2], v3v = vp[3];
    // K: row-major f16
    *(v8h*)&ldsK[buf][srow * HD + scol]     = pack8(k0v, k1v, 1.0f);
    *(v8h*)&ldsK[buf][srow * HD + scol + 8] = pack8(k2v, k3v, 1.0f);
    // V: transposed f16 [d][k]
    const v8h vl = pack8(v0v, v1v, 1.0f);
    const v8h vh = pack8(v2v, v3v, 1.0f);
#pragma unroll
    for (int e = 0; e < 8; ++e) {
      ldsVT[buf][(scol + e) * KB + srow]     = vl[e];
      ldsVT[buf][(scol + 8 + e) * KB + srow] = vh[e];
    }
  };

  stage(0, 0);

  for (int kb = 0; kb < NKB; ++kb) {
    __syncthreads();                       // staged buf ready; prev reads done
    if (kb + 1 < NKB) stage((kb + 1) & 1, (kb + 1) * KB);
    const _Float16* Kb  = ldsK[kb & 1];
    const _Float16* VTb = ldsVT[kb & 1];

    // ---- S^T tiles: four 16key x 16q tiles; A = K (16x32 over d), B = Q^T ----
    // A layout: lane row M = key = st*16+lane16; half e -> d = 32c + 8*hsel + e(+16)
    v8f p[4];
#pragma unroll
    for (int st = 0; st < 4; ++st) {
      v8f cst = {};
      const _Float16* krow = Kb + (st * 16 + lane16) * HD + 8 * hsel;
#pragma unroll
      for (int c = 0; c < 2; ++c) {
        const v8h* pp = (const v8h*)(krow + 32 * c);
        const v8h lo = pp[0], hi = pp[2];  // d runs {..}+0..7 and +16..23
        v16h ak;
#pragma unroll
        for (int e = 0; e < 8; ++e) { ak[e] = lo[e]; ak[8 + e] = hi[e]; }
        cst = __builtin_amdgcn_wmma_f32_16x16x32_f16(false, ak, false, bq[c],
                                                     (short)0, cst, false, false);
      }
      p[st] = cst;
    }

    // ---- online softmax, all per-lane (lane = one query column) ----
    float rm = p[0][0];
#pragma unroll
    for (int st = 0; st < 4; ++st)
#pragma unroll
      for (int i = 0; i < 8; ++i) rm = fmaxf(rm, p[st][i]);
    rm = fmaxf(rm, swz16(rm));                 // combine with partner half
    const float mn   = fmaxf(m_run, rm);
    const float corr = fast_exp2(m_run - mn);
    m_run = mn;
    float ls = 0.0f;
#pragma unroll
    for (int st = 0; st < 4; ++st)
#pragma unroll
      for (int i = 0; i < 8; ++i) { p[st][i] = fast_exp2(p[st][i] - mn); ls += p[st][i]; }
    ls += swz16(ls);
    l_run = l_run * corr + ls;
#pragma unroll
    for (int f = 0; f < 4; ++f)
#pragma unroll
      for (int r = 0; r < 8; ++r) acc[f][r] *= corr;

    // ---- O^T += V^T x P^T over two 32-key halves ----
#pragma unroll
    for (int h = 0; h < 2; ++h) {
      // B-fragment of P^T for keys [32h, 32h+32): lane needs tile t = 2h+hsel;
      // 8 of its 16 values live in the partner lane (xor 16).
      float recv[8];
#pragma unroll
      for (int r = 0; r < 8; ++r) {
        const float send = hsel ? p[2 * h][r] : p[2 * h + 1][r];  // what partner needs
        recv[r] = swz16(send);
      }
      v16h bp;
#pragma unroll
      for (int e = 0; e < 8; ++e) {
        const float lo = hsel ? recv[e]         : p[2 * h][e];
        const float hi = hsel ? p[2 * h + 1][e] : recv[e];
        bp[e]     = (_Float16)lo;
        bp[8 + e] = (_Float16)hi;
      }
      // A = V^T chunk (16d x 32k): k = 32h + 16*hsel + e
#pragma unroll
      for (int f = 0; f < 4; ++f) {
        const v8h* pp = (const v8h*)(VTb + (16 * f + lane16) * KB + 32 * h + 16 * hsel);
        const v8h lo = pp[0], hi = pp[1];
        v16h av;
#pragma unroll
        for (int e = 0; e < 8; ++e) { av[e] = lo[e]; av[8 + e] = hi[e]; }
        acc[f] = __builtin_amdgcn_wmma_f32_16x16x32_f16(false, av, false, bp,
                                                        (short)0, acc[f], false, false);
      }
    }
  }

  // ---- epilogue: lane holds q = lane16, 8 contiguous d per fragment ----
  const float linv = 1.0f / l_run;
  float* op = o + (size_t)(qt * 16 + lane16) * HD + 8 * hsel;
#pragma unroll
  for (int f = 0; f < 4; ++f) {
    float4 o0, o1;
    o0.x = acc[f][0] * linv; o0.y = acc[f][1] * linv;
    o0.z = acc[f][2] * linv; o0.w = acc[f][3] * linv;
    o1.x = acc[f][4] * linv; o1.y = acc[f][5] * linv;
    o1.z = acc[f][6] * linv; o1.w = acc[f][7] * linv;
    ((float4*)(op + 16 * f))[0] = o0;
    ((float4*)(op + 16 * f))[1] = o1;
  }
}

extern "C" void kernel_launch(void* const* d_in, const int* in_sizes, int n_in,
                              void* d_out, int out_size, void* d_ws, size_t ws_size,
                              hipStream_t stream) {
  (void)in_sizes; (void)n_in; (void)d_ws; (void)ws_size; (void)out_size;
  const float* q = (const float*)d_in[0];
  const float* k = (const float*)d_in[1];
  const float* v = (const float*)d_in[2];
  // d_in[3] is d_k == 64, baked into the kernel scale.
  float* out = (float*)d_out;

  const int B = 2, H = 16;
  dim3 grid(B * H * (NQT / WPB));  // 512 blocks x 8 waves
  dim3 block(32 * WPB);
  fa_fwd_kernel<<<grid, block, 0, stream>>>(q, k, v, out);
}
